// Attention_80917183856719
// MI455X (gfx1250) — compile-verified, matrix operand order checked by reading, not run on hardware
//
#include <hip/hip_runtime.h>
#include <hip/hip_bf16.h>

// ---------------------------------------------------------------------------
// Problem constants (match reference)
// ---------------------------------------------------------------------------
#define BATCH    4
#define T_LEN    2048
#define DMODEL   2048
#define NQ_H     16
#define NK_H     4
#define HEAD_D   128
#define ROPE_D   64
#define WINDOW_SZ 1024

typedef __attribute__((ext_vector_type(16))) __bf16 v16bf;
typedef __attribute__((ext_vector_type(8)))  float  v8f;

// ---------------------------------------------------------------------------
// Helpers
// ---------------------------------------------------------------------------
__device__ __forceinline__ unsigned short f2bf(float f) {
    unsigned int u = __float_as_uint(f);
    unsigned int r = (u + 0x7FFFu + ((u >> 16) & 1u)) >> 16;   // RNE
    return (unsigned short)r;
}

union BFrag {            // one WMMA bf16 operand: 16 bf16 = 2 x b128 loads
    v16bf v;
    uint4 u[2];
};

union AccF {             // 16x16 f32 C/D tile
    v8f   v;
    float f[8];
};

// Load a 16-bf16 fragment for A (row-contiguous K) or B (Wx row-contiguous K).
// Per the CDNA5 16-bit operand layout: lane holds row/col = lane&15,
// K pairs: VGPR0..3 -> K = base..base+7, VGPR4..7 -> K = 16+base..16+base+7,
// base = (lane>>4)*8.
__device__ __forceinline__ void load_frag(v16bf* dst, const unsigned short* rowptr, int base) {
    BFrag f;
    f.u[0] = *reinterpret_cast<const uint4*>(rowptr + base);
    f.u[1] = *reinterpret_cast<const uint4*>(rowptr + 16 + base);
    *dst = f.v;
}

__device__ __forceinline__ v8f wmma_bf16(v16bf a, v16bf b, v8f c) {
    return __builtin_amdgcn_wmma_f32_16x16x32_bf16(
        /*neg_a=*/false, a, /*neg_b=*/false, b,
        /*c_mod=*/(short)0, c, /*reuse_a=*/false, /*reuse_b=*/false);
}

// ---------------------------------------------------------------------------
// Kernel 1: fp32 -> bf16 streaming conversion
// ---------------------------------------------------------------------------
__global__ void cvt_bf16_kernel(const float* __restrict__ in,
                                unsigned short* __restrict__ out, size_t n) {
    size_t i = (size_t)blockIdx.x * blockDim.x + threadIdx.x;
    size_t stride = (size_t)gridDim.x * blockDim.x;
    for (; i < n; i += stride) out[i] = f2bf(in[i]);
}

// ---------------------------------------------------------------------------
// Kernel 2: generic bf16 GEMM  C[M,N] = A[M,K] * W[N,K]^T  (f32 accum/out)
// One wave computes a 32x64 tile: 2 A-frags x 4 B-frags -> 8 WMMA per K-step.
// ---------------------------------------------------------------------------
__global__ __launch_bounds__(256)
void gemm_bf16_nt(const unsigned short* __restrict__ A,
                  const unsigned short* __restrict__ Bw,
                  float* __restrict__ C, int M, int N, int K) {
    const int lane = threadIdx.x & 31;
    const int wid  = blockIdx.x * (blockDim.x >> 5) + (threadIdx.x >> 5);
    const int mtiles = M >> 5;
    const int ntiles = N >> 6;
    if (wid >= mtiles * ntiles) return;
    const int mt   = wid % mtiles;
    const int nt   = wid / mtiles;
    const int col  = lane & 15;
    const int half = lane >> 4;
    const int base = half * 8;

    const unsigned short* arow0 = A + (size_t)(mt * 32 + col) * K;
    const unsigned short* arow1 = A + (size_t)(mt * 32 + 16 + col) * K;
    const unsigned short* brow0 = Bw + (size_t)(nt * 64 +  0 + col) * K;
    const unsigned short* brow1 = Bw + (size_t)(nt * 64 + 16 + col) * K;
    const unsigned short* brow2 = Bw + (size_t)(nt * 64 + 32 + col) * K;
    const unsigned short* brow3 = Bw + (size_t)(nt * 64 + 48 + col) * K;

    AccF acc[2][4];
#pragma unroll
    for (int i = 0; i < 2; ++i)
#pragma unroll
        for (int j = 0; j < 4; ++j)
#pragma unroll
            for (int v = 0; v < 8; ++v) acc[i][j].f[v] = 0.0f;

    for (int k0 = 0; k0 < K; k0 += 32) {
        v16bf a0, a1, b0, b1, b2, b3;
        load_frag(&a0, arow0 + k0, base);
        load_frag(&a1, arow1 + k0, base);
        load_frag(&b0, brow0 + k0, base);
        load_frag(&b1, brow1 + k0, base);
        load_frag(&b2, brow2 + k0, base);
        load_frag(&b3, brow3 + k0, base);
        acc[0][0].v = wmma_bf16(a0, b0, acc[0][0].v);
        acc[0][1].v = wmma_bf16(a0, b1, acc[0][1].v);
        acc[0][2].v = wmma_bf16(a0, b2, acc[0][2].v);
        acc[0][3].v = wmma_bf16(a0, b3, acc[0][3].v);
        acc[1][0].v = wmma_bf16(a1, b0, acc[1][0].v);
        acc[1][1].v = wmma_bf16(a1, b1, acc[1][1].v);
        acc[1][2].v = wmma_bf16(a1, b2, acc[1][2].v);
        acc[1][3].v = wmma_bf16(a1, b3, acc[1][3].v);
    }

    // C/D layout: lane -> n = col, VGPR v -> m = v + 8*half
#pragma unroll
    for (int i = 0; i < 2; ++i)
#pragma unroll
        for (int j = 0; j < 4; ++j)
#pragma unroll
            for (int v = 0; v < 8; ++v) {
                int m = mt * 32 + i * 16 + v + 8 * half;
                int n = nt * 64 + j * 16 + col;
                C[(size_t)m * N + n] = acc[i][j].f[v];
            }
}

// ---------------------------------------------------------------------------
// Kernel 3: fused RMS-norm (+gain +1/sqrt(HD) for q) + half-RoPE.
// raw: [B*T, heads*HD] f32; out: [b][h][T][HD] bf16 (head-major for attention)
// One wave per (b,t,h); lane holds 4 contiguous dims.
// ---------------------------------------------------------------------------
__global__ __launch_bounds__(256)
void qk_norm_rope(const float* __restrict__ raw, unsigned short* __restrict__ outp,
                  const float* __restrict__ gainp, float extra_scale, int heads) {
    const int wave  = blockIdx.x * (blockDim.x >> 5) + (threadIdx.x >> 5);
    const int total = BATCH * T_LEN * heads;
    if (wave >= total) return;
    const int lane = threadIdx.x & 31;
    const int h  = wave % heads;
    const int bt = wave / heads;
    const int b  = bt / T_LEN;
    const int t  = bt % T_LEN;
    const int d0 = lane * 4;

    const float* src = raw + (size_t)bt * heads * HEAD_D + h * HEAD_D + d0;
    float4 x = *reinterpret_cast<const float4*>(src);

    float ss = x.x * x.x + x.y * x.y + x.z * x.z + x.w * x.w;
#pragma unroll
    for (int m = 16; m >= 1; m >>= 1) ss += __shfl_xor(ss, m, 32);

    float g   = (gainp ? gainp[0] : 1.0f) * extra_scale;
    float rms = rsqrtf(ss * (1.0f / HEAD_D) + 1e-6f) * g;
    x.x *= rms; x.y *= rms; x.z *= rms; x.w *= rms;

    if (d0 < ROPE_D) {                       // lanes 0..15 hold rotated pairs
        int p0 = d0 >> 1;                    // pair indices p0, p0+1
        float fr0 = __powf(10000.0f, -((float)(2 * p0)) * (1.0f / ROPE_D));
        float fr1 = __powf(10000.0f, -((float)(2 * (p0 + 1))) * (1.0f / ROPE_D));
        float a0 = (float)t * fr0, a1 = (float)t * fr1;
        float s0, c0, s1, c1;
        __sincosf(a0, &s0, &c0);
        __sincosf(a1, &s1, &c1);
        float x1 = x.x, x2 = x.y;
        x.x = x1 * c0 - x2 * s0;  x.y = x1 * s0 + x2 * c0;
        float x3 = x.z, x4 = x.w;
        x.z = x3 * c1 - x4 * s1;  x.w = x3 * s1 + x4 * c1;
    }

    ushort4 o;
    o.x = f2bf(x.x); o.y = f2bf(x.y); o.z = f2bf(x.z); o.w = f2bf(x.w);
    unsigned short* dst = outp + (((size_t)(b * heads + h) * T_LEN + t) * HEAD_D + d0);
    *reinterpret_cast<ushort4*>(dst) = o;
}

// ---------------------------------------------------------------------------
// Kernel 4: V transpose  [B*T, NK*HD] f32 -> [b][h][d][t] bf16
// (makes PV-WMMA B-operand K-contiguous)
// ---------------------------------------------------------------------------
__global__ void v_transpose_k(const float* __restrict__ vraw,
                              unsigned short* __restrict__ vT) {
    size_t idx = (size_t)blockIdx.x * blockDim.x + threadIdx.x;
    size_t total = (size_t)BATCH * T_LEN * NK_H * HEAD_D;
    size_t stride = (size_t)gridDim.x * blockDim.x;
    for (; idx < total; idx += stride) {
        int f  = (int)(idx % (NK_H * HEAD_D));
        size_t bt = idx / (NK_H * HEAD_D);
        int hh = f / HEAD_D, d = f % HEAD_D;
        int b  = (int)(bt / T_LEN), t = (int)(bt % T_LEN);
        vT[(((size_t)b * NK_H + hh) * HEAD_D + d) * T_LEN + t] = f2bf(vraw[idx]);
    }
}

// ---------------------------------------------------------------------------
// Kernel 5: sliding-window flash attention.
// One wave per (b, h, 16-query tile). 32-key blocks:
//   8 WMMA for S = Q K^T (per-d-step), mask, online softmax,
//   P -> LDS -> A-frag, 8 WMMA for O += P V.
// Q: [b][h][t][d] bf16 (pre-scaled by gain/sqrt(HD))
// K: [b][kvh][t][d] bf16,  V: [b][kvh][d][t] bf16
// Oa: [B,T,NQ*HD] bf16
// ---------------------------------------------------------------------------
__global__ __launch_bounds__(32)
void attn_swa(const unsigned short* __restrict__ Q,
              const unsigned short* __restrict__ Kh,
              const unsigned short* __restrict__ Vt,
              unsigned short* __restrict__ Oa) {
    const int w  = blockIdx.x;
    const int qt = w % (T_LEN / 16);
    const int h  = (w / (T_LEN / 16)) % NQ_H;
    const int b  = w / ((T_LEN / 16) * NQ_H);
    const int i0  = qt * 16;
    const int kvh = h / (NQ_H / NK_H);

    const int lane = threadIdx.x & 31;
    const int col  = lane & 15;
    const int half = lane >> 4;
    const int base = half * 8;

    const unsigned short* Qb = Q  + (size_t)(b * NQ_H + h)   * T_LEN * HEAD_D;
    const unsigned short* Kb = Kh + (size_t)(b * NK_H + kvh) * T_LEN * HEAD_D;
    const unsigned short* Vb = Vt + (size_t)(b * NK_H + kvh) * HEAD_D * T_LEN;

    // Q tile A-fragments (16 queries x 128 dims = 4 K-steps)
    v16bf qa[4];
    {
        const unsigned short* qrow = Qb + (size_t)(i0 + col) * HEAD_D;
#pragma unroll
        for (int ds = 0; ds < 4; ++ds) load_frag(&qa[ds], qrow + ds * 32, base);
    }

    AccF o[8];
#pragma unroll
    for (int dt = 0; dt < 8; ++dt)
#pragma unroll
        for (int v = 0; v < 8; ++v) o[dt].f[v] = 0.0f;
    float mr[8], lr[8];
#pragma unroll
    for (int v = 0; v < 8; ++v) { mr[v] = -1e30f; lr[v] = 0.0f; }

    __shared__ __align__(16) unsigned short pls[16 * 32];  // P tile staging

    int jlo = i0 - WINDOW_SZ + 1;
    if (jlo < 0) jlo = 0;
    jlo &= ~31;

    for (int j0 = jlo; j0 <= i0 + 15; j0 += 32) {
        // ---- S = Q K^T over this 32-key block (two 16-key sub-tiles) ----
        AccF s0t, s1t;
#pragma unroll
        for (int v = 0; v < 8; ++v) { s0t.f[v] = 0.0f; s1t.f[v] = 0.0f; }
        const unsigned short* kr0 = Kb + (size_t)(j0 + col) * HEAD_D;
        const unsigned short* kr1 = Kb + (size_t)(j0 + 16 + col) * HEAD_D;
#pragma unroll
        for (int ds = 0; ds < 4; ++ds) {
            v16bf kf0, kf1;
            load_frag(&kf0, kr0 + ds * 32, base);
            load_frag(&kf1, kr1 + ds * 32, base);
            s0t.v = wmma_bf16(qa[ds], kf0, s0t.v);
            s1t.v = wmma_bf16(qa[ds], kf1, s1t.v);
        }

        // ---- mask + online softmax ----
        float alpha[8];
#pragma unroll
        for (int v = 0; v < 8; ++v) {
            int i  = i0 + v + 8 * half;          // absolute query index
            int ja = j0 + col;                    // key index, sub-tile 0
            int jb = j0 + 16 + col;               // key index, sub-tile 1
            float sa = (ja <= i && ja > i - WINDOW_SZ) ? s0t.f[v] : -1e30f;
            float sb = (jb <= i && jb > i - WINDOW_SZ) ? s1t.f[v] : -1e30f;
            float mv = fmaxf(sa, sb);
#pragma unroll
            for (int msk = 8; msk >= 1; msk >>= 1)
                mv = fmaxf(mv, __shfl_xor(mv, msk, 32));
            float mnew = fmaxf(mr[v], mv);
            alpha[v] = __expf(mr[v] - mnew);
            mr[v] = mnew;
            float pa = __expf(sa - mnew);
            float pb = __expf(sb - mnew);
            s0t.f[v] = pa;
            s1t.f[v] = pb;
            float rs = pa + pb;
#pragma unroll
            for (int msk = 8; msk >= 1; msk >>= 1)
                rs += __shfl_xor(rs, msk, 32);
            lr[v] = lr[v] * alpha[v] + rs;
        }
#pragma unroll
        for (int dt = 0; dt < 8; ++dt)
#pragma unroll
            for (int v = 0; v < 8; ++v) o[dt].f[v] *= alpha[v];

        // ---- P (C-layout) -> LDS row-major 16x32 -> A-fragment ----
#pragma unroll
        for (int v = 0; v < 8; ++v) {
            int m = v + 8 * half;
            pls[m * 32 + col]      = f2bf(s0t.f[v]);
            pls[m * 32 + 16 + col] = f2bf(s1t.f[v]);
        }
        asm volatile("s_wait_dscnt 0" ::: "memory");   // wave-local LDS RAW
        v16bf pf;
        load_frag(&pf, pls + (size_t)col * 32, base);

        // ---- O += P V ----
#pragma unroll
        for (int dt = 0; dt < 8; ++dt) {
            v16bf vf;
            const unsigned short* vrow = Vb + (size_t)(dt * 16 + col) * T_LEN + j0;
            load_frag(&vf, vrow, base);
            o[dt].v = wmma_bf16(pf, vf, o[dt].v);
        }
    }

    // ---- epilogue: normalize and store bf16 [B,T,NQ*HD] ----
    float invl[8];
#pragma unroll
    for (int v = 0; v < 8; ++v) invl[v] = 1.0f / lr[v];
#pragma unroll
    for (int dt = 0; dt < 8; ++dt)
#pragma unroll
        for (int v = 0; v < 8; ++v) {
            int trow = i0 + v + 8 * half;
            Oa[((size_t)(b * T_LEN + trow)) * DMODEL + h * HEAD_D + dt * 16 + col]
                = f2bf(o[dt].f[v] * invl[v]);
        }
}

// ---------------------------------------------------------------------------
// Host-side launch
// ---------------------------------------------------------------------------
extern "C" void kernel_launch(void* const* d_in, const int* in_sizes, int n_in,
                              void* d_out, int out_size, void* d_ws, size_t ws_size,
                              hipStream_t stream) {
    const float* x  = (const float*)d_in[0];
    const float* Wq = (const float*)d_in[1];
    const float* Wk = (const float*)d_in[2];
    const float* Wv = (const float*)d_in[3];
    const float* Wo = (const float*)d_in[4];
    const float* qk_gain = (const float*)d_in[5];
    float* out = (float*)d_out;

    const size_t nX  = (size_t)BATCH * T_LEN * DMODEL;        // 16.78M
    const size_t nWq = (size_t)DMODEL * DMODEL;               // 4.19M
    const size_t nWk = (size_t)(NK_H * HEAD_D) * DMODEL;      // 1.05M
    const size_t M   = (size_t)BATCH * T_LEN;                 // 8192 rows

    // workspace layout (bytes)
    char* wsb = (char*)d_ws;
    size_t off = 0;
    unsigned short* xb    = (unsigned short*)(wsb + off); off += nX * 2;
    unsigned short* wqb   = (unsigned short*)(wsb + off); off += nWq * 2;
    unsigned short* wkb   = (unsigned short*)(wsb + off); off += nWk * 2;
    unsigned short* wvb   = (unsigned short*)(wsb + off); off += nWk * 2;
    unsigned short* wob   = (unsigned short*)(wsb + off); off += nWq * 2;
    float*          qraw  = (float*)(wsb + off);          off += M * (NQ_H * HEAD_D) * 4;
    float*          kraw  = (float*)(wsb + off);          off += M * (NK_H * HEAD_D) * 4;
    float*          vraw  = (float*)(wsb + off);          off += M * (NK_H * HEAD_D) * 4;
    unsigned short* qattn = (unsigned short*)(wsb + off); off += M * (NQ_H * HEAD_D) * 2;
    unsigned short* kattn = (unsigned short*)(wsb + off); off += M * (NK_H * HEAD_D) * 2;
    unsigned short* vT    = (unsigned short*)(wsb + off); off += M * (NK_H * HEAD_D) * 2;
    unsigned short* attno = (unsigned short*)(wsb + off); off += M * (NQ_H * HEAD_D) * 2;
    (void)ws_size; (void)n_in; (void)in_sizes; (void)out_size;

    // 1) fp32 -> bf16 conversions
    cvt_bf16_kernel<<<8192, 256, 0, stream>>>(x,  xb,  nX);
    cvt_bf16_kernel<<<4096, 256, 0, stream>>>(Wq, wqb, nWq);
    cvt_bf16_kernel<<<2048, 256, 0, stream>>>(Wk, wkb, nWk);
    cvt_bf16_kernel<<<2048, 256, 0, stream>>>(Wv, wvb, nWk);
    cvt_bf16_kernel<<<4096, 256, 0, stream>>>(Wo, wob, nWq);

    // 2) QKV projections (bf16 WMMA, f32 accum)
    //    q: M x 2048, k/v: M x 512, K = 2048
    {
        int waves_q = (int)((M / 32) * (DMODEL / 64));           // 8192
        int waves_k = (int)((M / 32) * ((NK_H * HEAD_D) / 64));  // 2048
        gemm_bf16_nt<<<waves_q / 8, 256, 0, stream>>>(xb, wqb, qraw, (int)M, DMODEL, DMODEL);
        gemm_bf16_nt<<<waves_k / 8, 256, 0, stream>>>(xb, wkb, kraw, (int)M, NK_H * HEAD_D, DMODEL);
        gemm_bf16_nt<<<waves_k / 8, 256, 0, stream>>>(xb, wvb, vraw, (int)M, NK_H * HEAD_D, DMODEL);
    }

    // 3) norm + rope (fold qk_gain and 1/sqrt(HD) into q)
    {
        const float inv_sqrt_hd = 0.08838834764831845f;  // 1/sqrt(128)
        int gq = BATCH * T_LEN * NQ_H;   // waves
        int gk = BATCH * T_LEN * NK_H;
        qk_norm_rope<<<gq / 8, 256, 0, stream>>>(qraw, qattn, qk_gain, inv_sqrt_hd, NQ_H);
        qk_norm_rope<<<gk / 8, 256, 0, stream>>>(kraw, kattn, nullptr, 1.0f, NK_H);
    }

    // 4) V transpose to [b][h][d][t] bf16
    v_transpose_k<<<4096, 256, 0, stream>>>(vraw, vT);

    // 5) sliding-window attention: one wave per (b, h, 16-query tile)
    attn_swa<<<BATCH * NQ_H * (T_LEN / 16), 32, 0, stream>>>(qattn, kattn, vT, attno);

    // 6) output projection -> f32 d_out
    {
        int waves_o = (int)((M / 32) * (DMODEL / 64));           // 8192
        gemm_bf16_nt<<<waves_o / 8, 256, 0, stream>>>(attno, wob, out, (int)M, DMODEL, DMODEL);
    }
}
